// SwinTransformerLayer_53472342835265
// MI455X (gfx1250) — matrix operand order repaired and checked by reading, not compiled, not run on hardware
//
#include <hip/hip_runtime.h>
#include <hip/hip_bf16.h>
#include <math.h>

typedef _Float16 v8h  __attribute__((ext_vector_type(8)));
typedef _Float16 v16h __attribute__((ext_vector_type(16)));
typedef float    v8f  __attribute__((ext_vector_type(8)));
typedef int      i32x4v __attribute__((vector_size(16)));   // matches builtin param type

#define BB     4
#define CC     192
#define HH     192
#define WWD    192
#define HEADS  6
#define HD     32
#define WIN    8
#define SHIFT  4
#define NWH    24
#define NWIN   576          // 24*24 windows per image
#define W2     64           // tokens per window
#define NTOK   (BB*NWIN*W2) // 147456 tokens
#define EMB    256
#define HID    768
#define SPAT   (HH*WWD)     // 36864

// ---- CDNA5 async global->LDS path (ASYNCcnt), with safe fallback ----
#define GPTR(p) ((__attribute__((address_space(1))) i32x4v*)(p))
#define LPTR(p) ((__attribute__((address_space(3))) i32x4v*)(p))
#if defined(__has_builtin)
#if __has_builtin(__builtin_amdgcn_global_load_async_to_lds_b128)
#define HAVE_ASYNC_LDS 1
#define ASYNC_LOAD_B128(g, l) __builtin_amdgcn_global_load_async_to_lds_b128((g), (l), 0, 0)
#endif
#endif
#ifndef HAVE_ASYNC_LDS
#define HAVE_ASYNC_LDS 0
#endif

#if defined(__has_builtin)
#if __has_builtin(__builtin_amdgcn_s_wait_asynccnt)
#define WAIT_ASYNC(n) __builtin_amdgcn_s_wait_asynccnt(n)
#endif
#endif
#ifndef WAIT_ASYNC
#define WAIT_ASYNC(n) asm volatile("s_wait_asynccnt %0" :: "i"(n) : "memory")
#endif

static __device__ __forceinline__ v8f vzero8f() {
  v8f z;
#pragma unroll
  for (int i = 0; i < 8; ++i) z[i] = 0.0f;
  return z;
}

static __device__ __forceinline__ int regid(int p) {
  // pre-shift region id along one axis (boundaries at H-WIN and H-SHIFT)
  return p < (HH - WIN) ? 0 : (p < (HH - SHIFT) ? 1 : 2);
}

// ---------------------------------------------------------------------------
// fp32 -> fp16 weight prepack (weights are [N,K] output-major already)
// ---------------------------------------------------------------------------
__global__ void f32_to_f16_kernel(const float* __restrict__ src,
                                  _Float16* __restrict__ dst, int n) {
  const int i = blockIdx.x * 256 + threadIdx.x;
  if (i < n) dst[i] = (_Float16)src[i];
}

// ---------------------------------------------------------------------------
// FiLM: gb[b, 0:2C] = cond[b] @ film_w^T + film_b   (tiny: 4 x 384)
// ---------------------------------------------------------------------------
__global__ __launch_bounds__(2*CC) void film_kernel(
    const float* __restrict__ cond, const float* __restrict__ fw,
    const float* __restrict__ fb, float* __restrict__ gb) {
  const int b = blockIdx.x, o = threadIdx.x;
  float acc = fb[o];
  const float* cp = cond + b * EMB;
  const float* wp = fw + (size_t)o * EMB;
  for (int k = 0; k < EMB; ++k) acc += cp[k] * wp[k];
  gb[b * 2 * CC + o] = acc;
}

// ---------------------------------------------------------------------------
// InstanceNorm per (b,c); mode 0: write shifted-window token layout (fuses
// cyclic shift + window partition); mode 1: apply FiLM, plain token layout.
// ---------------------------------------------------------------------------
__global__ __launch_bounds__(256) void instnorm_kernel(
    const float* __restrict__ x, _Float16* __restrict__ out,
    const float* __restrict__ gb, int mode) {
  __shared__ float rs[256];
  __shared__ float rq[256];
  __shared__ float sh[2];
  const int tid = threadIdx.x;
  const int bc  = blockIdx.x;
  const int b   = bc / CC, c = bc % CC;
  const float* xp = x + (size_t)bc * SPAT;

  float s0 = 0.f, s1 = 0.f;
  for (int i = tid; i < SPAT; i += 256) { float v = xp[i]; s0 += v; s1 += v * v; }
  rs[tid] = s0; rq[tid] = s1;
  __syncthreads();
  for (int st = 128; st > 0; st >>= 1) {
    if (tid < st) { rs[tid] += rs[tid + st]; rq[tid] += rq[tid + st]; }
    __syncthreads();
  }
  if (tid == 0) {
    const float mean = rs[0] / SPAT;
    const float var  = rq[0] / SPAT - mean * mean;
    const float inv  = rsqrtf(var + 1e-5f);
    float scale = inv, shift = -mean * inv;
    if (mode == 1) {
      const float g  = gb[b * 2 * CC + c];
      const float bt = gb[b * 2 * CC + CC + c];
      scale = inv * g; shift = bt - mean * inv * g;
    }
    sh[0] = scale; sh[1] = shift;
  }
  __syncthreads();
  const float scale = sh[0], shift = sh[1];

  for (int i = tid; i < SPAT; i += 256) {
    const float v = xp[i] * scale + shift;
    int tok;
    if (mode == 0) {
      const int hh = i / WWD, ww = i % WWD;
      const int hp = (hh + HH - SHIFT) % HH;   // roll by -SHIFT
      const int wp = (ww + WWD - SHIFT) % WWD;
      tok = (b * NWIN + (hp >> 3) * NWH + (wp >> 3)) * W2 + (hp & 7) * WIN + (wp & 7);
    } else {
      tok = b * SPAT + i;
    }
    out[(size_t)tok * CC + c] = (_Float16)v;
  }
}

// ---------------------------------------------------------------------------
// WMMA GEMM: out[M,N] = A[M,K](f16) * W[N,K]^T(f16) + bias, fp32 accumulate.
// 128 threads = 4 waves, 128x64 block tile (32x64 per wave -> 8 WMMA/K-step).
// Double-buffered async cache->LDS staging: issue slab k+1 (6 async ops per
// wave), s_wait_asynccnt 6 drains only slab k, compute overlaps the DMA.
// mode 0: f16 store (QKV)      mode 2: GELU(tanh) + f16 store (MLP1)
// mode 1: proj -> x1 = x + o, scatter to BCHW with inverse shift
// mode 3: final -> out = x1 + z, scatter to BCHW
// ---------------------------------------------------------------------------
__global__ __launch_bounds__(128) void wmma_gemm_kernel(
    const _Float16* __restrict__ A, const _Float16* __restrict__ Wm,
    const float* __restrict__ bias, _Float16* __restrict__ outH,
    float* __restrict__ outF, const float* __restrict__ resid,
    int M, int N, int K, int mode) {
  __shared__ __attribute__((aligned(16))) _Float16 As[2][128 * 32];
  __shared__ __attribute__((aligned(16))) _Float16 Bs[2][64 * 32];
  const int tid  = threadIdx.x;
  const int wave = tid >> 5;
  const int lane = tid & 31;
  const int lrow = lane & 15;
  const int lhi  = lane >> 4;
  const int m0 = blockIdx.y * 128;
  const int n0 = blockIdx.x * 64;

  v8f acc[2][4];
#pragma unroll
  for (int mi = 0; mi < 2; ++mi)
#pragma unroll
    for (int j = 0; j < 4; ++j) acc[mi][j] = vzero8f();

#if HAVE_ASYNC_LDS
  // A slab 128x32 halfs = 512 16B chunks (4 async ops/wave);
  // B slab  64x32 halfs = 256 16B chunks (2 async ops/wave).
  auto issue_slab = [&](int k0, int buf) {
#pragma unroll
    for (int i = 0; i < 4; ++i) {
      const int chunk = tid + 128 * i;
      const int row = chunk >> 2;
      const int ko  = (chunk & 3) * 8;
      ASYNC_LOAD_B128(GPTR(A + (size_t)(m0 + row) * K + k0 + ko),
                      LPTR(&As[buf][row * 32 + ko]));
    }
#pragma unroll
    for (int i = 0; i < 2; ++i) {
      const int chunk = tid + 128 * i;
      const int row = chunk >> 2;
      const int ko  = (chunk & 3) * 8;
      ASYNC_LOAD_B128(GPTR(Wm + (size_t)(n0 + row) * K + k0 + ko),
                      LPTR(&Bs[buf][row * 32 + ko]));
    }
  };
  issue_slab(0, 0);
#endif

  int kb = 0;
  for (int k0 = 0; k0 < K; k0 += 32, kb ^= 1) {
#if HAVE_ASYNC_LDS
    if (k0 + 32 < K) {
      issue_slab(k0 + 32, kb ^ 1);
      if (k0 + 64 < K)   // warm L2 two slabs ahead
        __builtin_prefetch(A + (size_t)(m0 + tid) * K + k0 + 64, 0, 1);
      WAIT_ASYNC(6);     // drain slab k only; slab k+1 stays in flight
    } else {
      WAIT_ASYNC(0);
    }
    __syncthreads();
#else
    // manual staging via VGPRs (b128 in / b128 out), single buffer
    {
      const v8h* sa = (const v8h*)(A + (size_t)(m0 + tid) * K + k0);
      v8h* da = (v8h*)(&As[0][tid * 32]);
#pragma unroll
      for (int i = 0; i < 4; ++i) da[i] = sa[i];
      const int br = tid >> 1, bko = (tid & 1) * 16;
      const v8h* sb = (const v8h*)(Wm + (size_t)(n0 + br) * K + k0 + bko);
      v8h* db = (v8h*)(&Bs[0][br * 32 + bko]);
      db[0] = sb[0]; db[1] = sb[1];
    }
    if (k0 + 32 < K)
      __builtin_prefetch(A + (size_t)(m0 + tid) * K + k0 + 32, 0, 1);
    __syncthreads();
#endif

#if HAVE_ASYNC_LDS
    const _Float16* Asb = &As[kb][0];
    const _Float16* Bsb = &Bs[kb][0];
#else
    const _Float16* Asb = &As[0][0];
    const _Float16* Bsb = &Bs[0][0];
#endif

    // B fragments once per K-step (row N, K halves 8h..8h+7 and 16+8h..16+8h+7 per lane)
    union { v16h v; v8h h[2]; } ub[4];
#pragma unroll
    for (int j = 0; j < 4; ++j) {
      const int bn = j * 16 + lrow;
      ub[j].h[0] = *(const v8h*)(Bsb + bn * 32 + lhi * 8);
      ub[j].h[1] = *(const v8h*)(Bsb + bn * 32 + lhi * 8 + 16);
    }
    // two A row-tiles per wave, 4 WMMAs each
#pragma unroll
    for (int mi = 0; mi < 2; ++mi) {
      union { v16h v; v8h h[2]; } ua;
      const int am = wave * 32 + mi * 16 + lrow;
      ua.h[0] = *(const v8h*)(Asb + am * 32 + lhi * 8);
      ua.h[1] = *(const v8h*)(Asb + am * 32 + lhi * 8 + 16);
#pragma unroll
      for (int j = 0; j < 4; ++j)
        acc[mi][j] = __builtin_amdgcn_wmma_f32_16x16x32_f16(
            false, ua.v, false, ub[j].v, (short)0, acc[mi][j], false, false);
    }
    __syncthreads();
  }

#pragma unroll
  for (int mi = 0; mi < 2; ++mi)
#pragma unroll
    for (int j = 0; j < 4; ++j) {
      const int n = n0 + j * 16 + lrow;
      const float bv = bias[n];
#pragma unroll
      for (int r = 0; r < 8; ++r) {
        const int m = m0 + wave * 32 + mi * 16 + r + lhi * 8;
        float v = acc[mi][j][r] + bv;
        if (mode == 0) {
          outH[(size_t)m * N + n] = (_Float16)v;
        } else if (mode == 2) {
          const float t = tanhf(0.7978845608f * (v + 0.044715f * v * v * v));
          outH[(size_t)m * N + n] = (_Float16)(0.5f * v * (1.0f + t));
        } else if (mode == 1) {
          const int bb  = m / (NWIN * W2);
          const int rem = m % (NWIN * W2);
          const int win = rem >> 6;
          const int tt  = rem & 63;
          const int hh = ((win / NWH) * WIN + (tt >> 3) + SHIFT) % HH;   // roll +s
          const int ww = ((win % NWH) * WIN + (tt & 7) + SHIFT) % WWD;
          const size_t xi = (((size_t)bb * CC + n) * HH + hh) * WWD + ww;
          outF[xi] = resid[xi] + v;
        } else {
          const int bb = m / SPAT;
          const int sp = m % SPAT;
          const size_t xi = (((size_t)bb * CC + n) * HH + (sp / WWD)) * WWD + (sp % WWD);
          outF[xi] = resid[xi] + v;
        }
      }
    }
}

// ---------------------------------------------------------------------------
// Shifted-window attention, one block (4 waves) per (window, head).
// S = QK^T (WMMA) -> +rel-pos bias +shift mask -> softmax (lane-group
// reductions) -> P restaged via LDS into A layout -> O = PV (WMMA).
// ---------------------------------------------------------------------------
__global__ __launch_bounds__(128) void swin_attn_kernel(
    const _Float16* __restrict__ qkv, const float* __restrict__ rpb,
    _Float16* __restrict__ otok) {
  __shared__ __attribute__((aligned(16))) _Float16 kbuf[64 * 32];  // [tok][d]
  __shared__ __attribute__((aligned(16))) _Float16 vbuf[32 * 64];  // [d][tok]
  __shared__ __attribute__((aligned(16))) _Float16 pbuf[4][16 * 64];

  const int tid  = threadIdx.x;
  const int wave = tid >> 5;
  const int lane = tid & 31;
  const int lrow = lane & 15;
  const int lhi  = lane >> 4;

  const int head  = blockIdx.x % HEADS;
  const int bw    = blockIdx.x / HEADS;   // 0 .. B*nWin-1
  const int win   = bw % NWIN;
  const int wbase = bw * W2;

  // stage K row-major, V transposed
  {
    const int r = tid >> 1, ko = (tid & 1) * 16;
    const v8h* sk = (const v8h*)(qkv + (size_t)(wbase + r) * (3 * CC) + CC + head * HD + ko);
    v8h* dk = (v8h*)(kbuf + r * 32 + ko);
    dk[0] = sk[0]; dk[1] = sk[1];
  }
  for (int e = tid; e < HD * W2; e += 128) {
    const int d = e >> 6, tk = e & 63;
    vbuf[d * 64 + tk] = qkv[(size_t)(wbase + tk) * (3 * CC) + 2 * CC + head * HD + d];
  }
  __syncthreads();

  // Q fragment straight from global (row stride 3C halves, 16B aligned)
  union { v16h v; v8h h[2]; } ua;
  {
    const _Float16* qp = qkv + (size_t)(wbase + wave * 16 + lrow) * (3 * CC) + head * HD + lhi * 8;
    ua.h[0] = *(const v8h*)qp;
    ua.h[1] = *(const v8h*)(qp + 16);
  }

  // S tiles: 16 queries x 64 keys, K-dim = hd = 32 (single WMMA each)
  v8f s[4];
#pragma unroll
  for (int j = 0; j < 4; ++j) {
    union { v16h v; v8h h[2]; } ub;
    const int bn = j * 16 + lrow;
    ub.h[0] = *(const v8h*)(kbuf + bn * 32 + lhi * 8);
    ub.h[1] = *(const v8h*)(kbuf + bn * 32 + lhi * 8 + 16);
    s[j] = __builtin_amdgcn_wmma_f32_16x16x32_f16(
        false, ua.v, false, ub.v, (short)0, vzero8f(), false, false);
  }

  // scale + relative position bias + SW-MSA mask
  const int whh = win / NWH, wcc = win % NWH;
#pragma unroll
  for (int j = 0; j < 4; ++j) {
    const int kt = j * 16 + lrow;
    const int kr = kt >> 3, kc = kt & 7;
    const int kid = regid(whh * WIN + kr) * 3 + regid(wcc * WIN + kc);
#pragma unroll
    for (int r = 0; r < 8; ++r) {
      const int q  = wave * 16 + r + lhi * 8;
      const int qr = q >> 3, qc = q & 7;
      const int idx = (qr - kr + 7) * 15 + (qc - kc + 7);
      float val = s[j][r] * 0.1767766953f + rpb[idx * HEADS + head];
      const int qid = regid(whh * WIN + qr) * 3 + regid(wcc * WIN + qc);
      if (qid != kid) val -= 100.0f;
      s[j][r] = val;
    }
  }

  // softmax over 64 keys: per row, reduce across lane-group of 16 + 4 tiles
#pragma unroll
  for (int r = 0; r < 8; ++r) {
    float mx = fmaxf(fmaxf(s[0][r], s[1][r]), fmaxf(s[2][r], s[3][r]));
#pragma unroll
    for (int off = 1; off < 16; off <<= 1) mx = fmaxf(mx, __shfl_xor(mx, off));
    float sum = 0.0f;
#pragma unroll
    for (int j = 0; j < 4; ++j) { s[j][r] = __expf(s[j][r] - mx); sum += s[j][r]; }
#pragma unroll
    for (int off = 1; off < 16; off <<= 1) sum += __shfl_xor(sum, off);
    const float inv = 1.0f / sum;
    const int prow = (r + lhi * 8) * 64;
#pragma unroll
    for (int j = 0; j < 4; ++j) {
      s[j][r] *= inv;
      pbuf[wave][prow + j * 16 + lrow] = (_Float16)s[j][r];
    }
  }
  __syncthreads();

  // O = P @ V : K-dim 64 split in two 32-chunks; N-dim 32 = two tiles
  v8f o[2];
  o[0] = vzero8f(); o[1] = vzero8f();
#pragma unroll
  for (int c0 = 0; c0 < 64; c0 += 32) {
    union { v16h v; v8h h[2]; } pa;
    const _Float16* pp = &pbuf[wave][lrow * 64 + c0 + lhi * 8];
    pa.h[0] = *(const v8h*)pp;
    pa.h[1] = *(const v8h*)(pp + 16);
#pragma unroll
    for (int j2 = 0; j2 < 2; ++j2) {
      union { v16h v; v8h h[2]; } vb;
      const _Float16* vp = vbuf + (j2 * 16 + lrow) * 64 + c0 + lhi * 8;
      vb.h[0] = *(const v8h*)vp;
      vb.h[1] = *(const v8h*)(vp + 16);
      o[j2] = __builtin_amdgcn_wmma_f32_16x16x32_f16(
          false, pa.v, false, vb.v, (short)0, o[j2], false, false);
    }
  }
#pragma unroll
  for (int j2 = 0; j2 < 2; ++j2)
#pragma unroll
    for (int r = 0; r < 8; ++r) {
      const int tok = wbase + wave * 16 + r + lhi * 8;
      otok[(size_t)tok * CC + head * HD + j2 * 16 + lrow] = (_Float16)o[j2][r];
    }
}

// ---------------------------------------------------------------------------
extern "C" void kernel_launch(void* const* d_in, const int* in_sizes, int n_in,
                              void* d_out, int out_size, void* d_ws, size_t ws_size,
                              hipStream_t stream) {
  const float* x      = (const float*)d_in[0];
  const float* cond   = (const float*)d_in[1];
  const float* qkv_w  = (const float*)d_in[2];
  const float* qkv_b  = (const float*)d_in[3];
  const float* proj_w = (const float*)d_in[4];
  const float* proj_b = (const float*)d_in[5];
  const float* rpb    = (const float*)d_in[6];
  const float* film_w = (const float*)d_in[7];
  const float* film_b = (const float*)d_in[8];
  const float* w1     = (const float*)d_in[9];
  const float* b1     = (const float*)d_in[10];
  const float* w2     = (const float*)d_in[11];
  const float* b2     = (const float*)d_in[12];

  char* ws = (char*)d_ws;
  size_t off = 0;
  auto carve = [&](size_t bytes) -> void* {
    void* p = (void*)(ws + off);
    off += (bytes + 255) & ~(size_t)255;
    return p;
  };
  _Float16* qkv_wh  = (_Float16*)carve((size_t)3 * CC * CC * 2);
  _Float16* proj_wh = (_Float16*)carve((size_t)CC * CC * 2);
  _Float16* w1h     = (_Float16*)carve((size_t)HID * CC * 2);
  _Float16* w2h     = (_Float16*)carve((size_t)CC * HID * 2);
  float*    gb      = (float*)carve((size_t)BB * 2 * CC * 4);
  _Float16* ytok    = (_Float16*)carve((size_t)NTOK * CC * 2);      // reused as ztok
  _Float16* qkvbuf  = (_Float16*)carve((size_t)NTOK * 3 * CC * 2);  // reused by h1
  _Float16* otok    = (_Float16*)carve((size_t)NTOK * CC * 2);      // reused by h1
  float*    x1      = (float*)carve((size_t)NTOK * CC * 4);
  _Float16* h1      = qkvbuf;   // MLP hidden overlaps dead qkv+otok region
  _Float16* ztok    = ytok;     // branch-2 tokens overlap dead branch-1 tokens

  // weight prepack + FiLM
  f32_to_f16_kernel<<<(3 * CC * CC + 255) / 256, 256, 0, stream>>>(qkv_w, qkv_wh, 3 * CC * CC);
  f32_to_f16_kernel<<<(CC * CC + 255) / 256, 256, 0, stream>>>(proj_w, proj_wh, CC * CC);
  f32_to_f16_kernel<<<(HID * CC + 255) / 256, 256, 0, stream>>>(w1, w1h, HID * CC);
  f32_to_f16_kernel<<<(CC * HID + 255) / 256, 256, 0, stream>>>(w2, w2h, CC * HID);
  film_kernel<<<BB, 2 * CC, 0, stream>>>(cond, film_w, film_b, gb);

  // branch 1: norm1 (fused shift+partition) -> QKV -> attention -> proj+residual
  instnorm_kernel<<<BB * CC, 256, 0, stream>>>(x, ytok, nullptr, 0);
  wmma_gemm_kernel<<<dim3((3 * CC) / 64, NTOK / 128), 128, 0, stream>>>(
      ytok, qkv_wh, qkv_b, qkvbuf, nullptr, nullptr, NTOK, 3 * CC, CC, 0);
  swin_attn_kernel<<<BB * NWIN * HEADS, 128, 0, stream>>>(qkvbuf, rpb, otok);
  wmma_gemm_kernel<<<dim3(CC / 64, NTOK / 128), 128, 0, stream>>>(
      otok, proj_wh, proj_b, nullptr, x1, x, NTOK, CC, CC, 1);

  // branch 2: norm2 + FiLM -> MLP1 (GELU) -> MLP2 + residual
  instnorm_kernel<<<BB * CC, 256, 0, stream>>>(x1, ztok, gb, 1);
  wmma_gemm_kernel<<<dim3(HID / 64, NTOK / 128), 128, 0, stream>>>(
      ztok, w1h, b1, h1, nullptr, nullptr, NTOK, HID, CC, 2);
  wmma_gemm_kernel<<<dim3(CC / 64, NTOK / 128), 128, 0, stream>>>(
      h1, w2h, b2, nullptr, (float*)d_out, x1, NTOK, CC, HID, 3);
}
